// LSTMWithAdapters_9552007266505
// MI455X (gfx1250) — matrix operand order repaired and checked by reading, not compiled, not run on hardware
//
#include <hip/hip_runtime.h>
#include <hip/hip_bf16.h>

// Problem sizes (fixed by the reference)
#define T_   512
#define B_   32
#define H_   1024
#define G4H  4096   // 4*H
#define DA_  64
#define NWG  32     // persistent workgroups in the scan kernel (32 j-columns each)

typedef __attribute__((ext_vector_type(16))) __bf16       v16bf;
typedef __attribute__((ext_vector_type(8)))  float        v8f;
typedef __attribute__((ext_vector_type(8)))  unsigned int u32x8;
typedef __attribute__((ext_vector_type(4)))  unsigned int u32x4;

// CDNA5 async global->LDS path (guarded: falls back to plain copies if the
// toolchain doesn't expose the builtins).
#if defined(__has_builtin)
#  if __has_builtin(__builtin_amdgcn_global_load_async_to_lds_b128) && \
      __has_builtin(__builtin_amdgcn_s_wait_asynccnt)
#    define HAVE_ASYNC_LDS 1
#  endif
#endif
#ifndef HAVE_ASYNC_LDS
#  define HAVE_ASYNC_LDS 0
#endif

// Exact parameter type reported by the compiler for the async-LDS builtin:
// '__attribute__((__vector_size__(4 * sizeof(int)))) int *'
typedef __attribute__((__vector_size__(4 * sizeof(int)))) int i32x4g;
typedef __attribute__((address_space(3))) void*   lds_vptr;
typedef __attribute__((address_space(3))) i32x4g* lds_i32x4_p;

// ---------- helpers ----------

__device__ __forceinline__ unsigned short f2bf(float f) {
  // round-to-nearest-even fp32 -> bf16
  unsigned u = __builtin_bit_cast(unsigned, f);
  u += 0x7FFFu + ((u >> 16) & 1u);
  return (unsigned short)(u >> 16);
}

// A-fragment (16x32 bf16, M x K) from row-major bf16 with leading dim ldk.
// ISA layout: lanes 0-15: M=lane, VGPR0-3 = K 0..7, VGPR4-7 = K 16..23;
//             lanes 16-31: M=lane-16, VGPR0-3 = K 8..15, VGPR4-7 = K 24..31.
__device__ __forceinline__ v16bf load_a_frag(const unsigned short* tile, int ldk, int lane) {
  int r   = lane & 15;
  int off = (lane & 16) >> 1;           // 0 or 8 bf16 elements
  const unsigned short* p = tile + (size_t)r * ldk + off;
  u32x4 lo = *(const u32x4*)p;          // 16B: K k0+0..7 (or 8..15)
  u32x4 hi = *(const u32x4*)(p + 16);   // 16B: K k0+16..23 (or 24..31)
  u32x8 v = { lo.x, lo.y, lo.z, lo.w, hi.x, hi.y, hi.z, hi.w };
  return __builtin_bit_cast(v16bf, v);
}

// B-fragment (32x16 bf16, K x N) from the pre-swizzled weight layout:
// frag[ntile][ktile][lane] = 8 dwords; lane holds col N=n0+(lane&15), dword v = {K=k0+8*(lane>>4)+2v+8*(v>>2), +1}.
__device__ __forceinline__ v16bf load_b_frag(const unsigned short* wsz, int ntile, int kt, int lane) {
  const u32x8* p = (const u32x8*)wsz;
  return __builtin_bit_cast(v16bf, p[((size_t)ntile * 32 + kt) * 32 + lane]);
}

__device__ __forceinline__ v8f wmma_bf16(v16bf a, v16bf b, v8f c) {
  return __builtin_amdgcn_wmma_f32_16x16x32_bf16(false, a, false, b, (short)0, c, false, false);
}

// ---------- kernels ----------

// fp32 -> bf16 elementwise
__global__ void k_f32_to_bf16(const float* __restrict__ in, unsigned short* __restrict__ out, int n) {
  int i = blockIdx.x * 256 + threadIdx.x;
  if (i < n) out[i] = f2bf(in[i]);
}

// gbias = b_ih + b_hh
__global__ void k_bias(const float* __restrict__ a, const float* __restrict__ b, float* __restrict__ o) {
  int i = blockIdx.x * 256 + threadIdx.x;
  if (i < G4H) o[i] = a[i] + b[i];
}

// Swizzle W [N=4096][K=1024] row-major fp32 into bf16 B-fragment layout.
// One thread per output dword (N*K/2 = 2,097,152 dwords).
__global__ void k_swizzle_w(const float* __restrict__ W, unsigned short* __restrict__ out) {
  unsigned dw = blockIdx.x * 256 + threadIdx.x;
  int v    = dw & 7;
  int lane = (dw >> 3) & 31;
  int kt   = (dw >> 8) & 31;     // KT = H/32 = 32
  int nt   = dw >> 13;           // NT = 4096/16 = 256
  int k = kt * 32 + ((lane >> 4) << 3) + 2 * v + ((v >> 2) << 3);
  int n = nt * 16 + (lane & 15);
  unsigned short lo = f2bf(W[(size_t)n * H_ + k]);
  unsigned short hi = f2bf(W[(size_t)n * H_ + k + 1]);
  ((unsigned*)out)[dw] = (unsigned)lo | ((unsigned)hi << 16);
}

// gx = x @ W_ih^T + (b_ih + b_hh):  M=T*B=16384, N=4096, K=1024, bf16 WMMA, fp32 acc.
// Grid (N/256, M/64); 256 threads = 8 waves in 2(M) x 4(N); wave tile 32x64 = 2x4 WMMA tiles.
__global__ __launch_bounds__(256) void k_gemm_gx(
    const unsigned short* __restrict__ xa, const unsigned short* __restrict__ wsz,
    const float* __restrict__ gbias, float* __restrict__ gx) {
  const int lane = threadIdx.x & 31;
  const int wv   = threadIdx.x >> 5;
  const int m0 = blockIdx.y * 64 + (wv >> 2) * 32;
  const int n0 = blockIdx.x * 256 + (wv & 3) * 64;
  v8f acc0[4], acc1[4];
  #pragma unroll
  for (int j = 0; j < 4; ++j)
    for (int e = 0; e < 8; ++e) { acc0[j][e] = 0.f; acc1[j][e] = 0.f; }

  for (int kt = 0; kt < 32; ++kt) {
    v16bf a0 = load_a_frag(xa + (size_t)m0 * H_ + kt * 32, H_, lane);
    v16bf a1 = load_a_frag(xa + (size_t)(m0 + 16) * H_ + kt * 32, H_, lane);
    #pragma unroll
    for (int j = 0; j < 4; ++j) {
      v16bf b = load_b_frag(wsz, (n0 >> 4) + j, kt, lane);
      acc0[j] = wmma_bf16(a0, b, acc0[j]);
      acc1[j] = wmma_bf16(a1, b, acc1[j]);
    }
  }
  // C layout: VGPR r -> M = r + 8*(lane>>4), N = lane&15
  const int nl = lane & 15;
  const int mh = (lane >> 4) << 3;
  #pragma unroll
  for (int j = 0; j < 4; ++j) {
    int n = n0 + j * 16 + nl;
    float bias = gbias[n];
    #pragma unroll
    for (int r = 0; r < 8; ++r) {
      gx[(size_t)(m0 + mh + r) * G4H + n]      = acc0[j][r] + bias;
      gx[(size_t)(m0 + 16 + mh + r) * G4H + n] = acc1[j][r] + bias;
    }
  }
}

// Persistent recurrent scan. NWG=32 workgroups; WG wg owns h/c columns [wg*32, wg*32+32)
// => gate columns {q*1024 + wg*32 + j} for q=0..3 (i,f,g,o), i.e. N=128 gate cols per WG.
// The wave's K=1024 slice of W_hh (32 B-fragments = 256 VGPRs) is held in REGISTERS for
// all 512 steps. Per step: async-stage h_{t-1} (bf16, 64KB) to LDS; preload gx gate
// values into registers; 8 waves each do one 32(M)x16(N) WMMA column; gates + c/h update
// with c resident in LDS; write h_t (bf16 ping-pong slab + fp32 activation); grid barrier.
__global__ __launch_bounds__(256, 1) void k_scan(
    const float* __restrict__ gx, const unsigned short* __restrict__ wsz,
    unsigned short* __restrict__ hbf /* 2 slabs of B_*H_ bf16 */,
    const float* __restrict__ c_in, float* __restrict__ act,
    float* __restrict__ h_out, float* __restrict__ c_out,
    unsigned* __restrict__ ctr) {
  extern __shared__ unsigned char smem[];
  unsigned short* h_lds = (unsigned short*)smem;                         // 32x1024 bf16 = 64KB
  float* g_lds = (float*)(smem + (size_t)B_ * H_ * 2);                   // 32x128 f32  = 16KB
  float* c_lds = (float*)(smem + (size_t)B_ * H_ * 2 + (size_t)B_ * 128 * 4); // 32x32 = 4KB

  const int tid  = threadIdx.x;
  const int lane = tid & 31;
  const int wv   = tid >> 5;
  const int wg   = blockIdx.x;
  const int j0   = wg * 32;

  for (int i = tid; i < B_ * 32; i += 256) {
    int b = i >> 5, j = i & 31;
    c_lds[i] = c_in[b * H_ + j0 + j];
  }

  const int q     = wv >> 1;                       // gate
  const int ntile = q * 64 + wg * 2 + (wv & 1);    // global 16-col tile in [0,256)

  // W_hh slice for this wave: resident in VGPRs for the whole scan.
  v16bf bfr[32];
  #pragma unroll
  for (int kt = 0; kt < 32; ++kt) bfr[kt] = load_b_frag(wsz, ntile, kt, lane);

  for (int t = 0; t < T_; ++t) {
    __syncthreads();
    // Preload this thread's gx gate pre-activations (in flight during the GEMM).
    float gxv[4][4];
    #pragma unroll
    for (int bb = 0; bb < 4; ++bb) {
      size_t row = ((size_t)t * B_ + ((wv << 2) + bb)) * G4H + j0 + lane;
      gxv[bb][0] = gx[row];
      gxv[bb][1] = gx[row + 1024];
      gxv[bb][2] = gx[row + 2048];
      gxv[bb][3] = gx[row + 3072];
    }
    { // stage h_{t-1} slab into LDS
      const unsigned short* src = hbf + (size_t)(t & 1) * B_ * H_;
#if HAVE_ASYNC_LDS
      for (int i = tid; i < (B_ * H_) / 8; i += 256) {
        __builtin_amdgcn_global_load_async_to_lds_b128(
            (i32x4g*)(src + (size_t)i * 8),
            (lds_i32x4_p)(lds_vptr)(void*)(h_lds + (size_t)i * 8),
            0, 0);
      }
      __builtin_amdgcn_s_wait_asynccnt(0);
#else
      const u32x4* s4 = (const u32x4*)src;
      u32x4* d4 = (u32x4*)h_lds;
      for (int i = tid; i < (B_ * H_) / 8; i += 256) d4[i] = s4[i];
#endif
    }
    __syncthreads();

    v8f acc0, acc1;
    for (int e = 0; e < 8; ++e) { acc0[e] = 0.f; acc1[e] = 0.f; }
    #pragma unroll
    for (int kt = 0; kt < 32; ++kt) {
      v16bf a0 = load_a_frag(h_lds + kt * 32, H_, lane);
      v16bf a1 = load_a_frag(h_lds + 16 * H_ + kt * 32, H_, lane);
      acc0 = wmma_bf16(a0, bfr[kt], acc0);
      acc1 = wmma_bf16(a1, bfr[kt], acc1);
    }
    // g_lds column = wv*16 + nl == q*32 + j  (j = (wv&1)*16 + nl)
    const int nl = lane & 15;
    const int mh = (lane >> 4) << 3;
    const int col = wv * 16 + nl;
    #pragma unroll
    for (int r = 0; r < 8; ++r) {
      g_lds[(mh + r) * 128 + col]      = acc0[r];
      g_lds[(16 + mh + r) * 128 + col] = acc1[r];
    }
    __syncthreads();

    unsigned short* hw = hbf + (size_t)((t + 1) & 1) * B_ * H_;
    #pragma unroll
    for (int bb = 0; bb < 4; ++bb) {
      int b = (wv << 2) + bb;
      int j = lane;
      size_t row = (size_t)t * B_ + b;
      float gi = gxv[bb][0] + g_lds[b * 128 +  0 + j];
      float gf = gxv[bb][1] + g_lds[b * 128 + 32 + j];
      float gg = gxv[bb][2] + g_lds[b * 128 + 64 + j];
      float go = gxv[bb][3] + g_lds[b * 128 + 96 + j];
      float si = 1.f / (1.f + __expf(-gi));
      float sf = 1.f / (1.f + __expf(-gf));
      float so = 1.f / (1.f + __expf(-go));
      float tg = tanhf(gg);
      float c = sf * c_lds[b * 32 + j] + si * tg;
      c_lds[b * 32 + j] = c;
      float h = so * tanhf(c);
      act[row * H_ + j0 + j] = h;          // layer output (pre-adapter), fp32
      hw[b * H_ + j0 + j] = f2bf(h);       // next-step GEMM operand, bf16
      if (t == T_ - 1) {
        h_out[b * H_ + j0 + j] = h;
        c_out[b * H_ + j0 + j] = c;
      }
    }
    __threadfence();
    __syncthreads();
    if (tid == 0) {
      atomicAdd(ctr, 1u);
      unsigned target = (unsigned)NWG * (unsigned)(t + 1);
      while (atomicAdd(ctr, 0u) < target) __builtin_amdgcn_s_sleep(8);
    }
    __syncthreads();
    __threadfence();
  }
}

// Fused adapter (x + up(relu(down(x)))) + LayerNorm, one 256-thread WG per row.
// Writes fp32 result and a bf16 copy (A operand for the next layer's input GEMM).
__global__ __launch_bounds__(256) void k_adapter_ln(
    const float* __restrict__ act, const float* __restrict__ Adw, const float* __restrict__ Adb,
    const float* __restrict__ Auw, const float* __restrict__ Aub,
    const float* __restrict__ lng, const float* __restrict__ lnb,
    float* __restrict__ xout, unsigned short* __restrict__ xa) {
  __shared__ float xrow[H_];
  __shared__ float dred[256];
  __shared__ float dvec[DA_];
  __shared__ float rsum[8], rssq[8], stats[2];
  const int tid = threadIdx.x;
  const int lane = tid & 31, wv = tid >> 5;
  const size_t row = blockIdx.x;
  const float* xr = act + row * H_;
  for (int i = tid; i < H_; i += 256) xrow[i] = xr[i];
  __syncthreads();
  { // down-projection: 4 threads per output a, 256-wide partial dots
    const int a = tid >> 2, qd = tid & 3;
    const float* wa = Adw + (size_t)a * H_ + qd * 256;
    const float* xq = xrow + qd * 256;
    float s = 0.f;
    for (int k = 0; k < 256; ++k) s = fmaf(wa[k], xq[k], s);
    dred[tid] = s;
  }
  __syncthreads();
  if ((tid & 3) == 0) {
    int a = tid >> 2;
    float v = dred[tid] + dred[tid + 1] + dred[tid + 2] + dred[tid + 3] + Adb[a];
    dvec[a] = v > 0.f ? v : 0.f;   // relu
  }
  __syncthreads();
  float o[4], s1 = 0.f, s2 = 0.f;
  #pragma unroll
  for (int i = 0; i < 4; ++i) {
    int h = tid + i * 256;
    const float* wu = Auw + (size_t)h * DA_;
    float a = xrow[h] + Aub[h];
    for (int aa = 0; aa < DA_; ++aa) a = fmaf(dvec[aa], wu[aa], a);
    o[i] = a; s1 += a; s2 += a * a;
  }
  for (int off = 16; off > 0; off >>= 1) { s1 += __shfl_down(s1, off); s2 += __shfl_down(s2, off); }
  if (lane == 0) { rsum[wv] = s1; rssq[wv] = s2; }
  __syncthreads();
  if (tid == 0) {
    float a = 0.f, b = 0.f;
    for (int i = 0; i < 8; ++i) { a += rsum[i]; b += rssq[i]; }
    float mean = a * (1.f / H_);
    float var  = b * (1.f / H_) - mean * mean;
    stats[0] = mean; stats[1] = rsqrtf(var + 1e-5f);
  }
  __syncthreads();
  float mean = stats[0], rstd = stats[1];
  #pragma unroll
  for (int i = 0; i < 4; ++i) {
    int h = tid + i * 256;
    float y = (o[i] - mean) * rstd * lng[h] + lnb[h];
    xout[row * H_ + h] = y;
    xa[row * H_ + h] = f2bf(y);
  }
}

// ---------- launcher ----------

extern "C" void kernel_launch(void* const* d_in, const int* in_sizes, int n_in,
                              void* d_out, int out_size, void* d_ws, size_t ws_size,
                              hipStream_t stream) {
  const float* x   = (const float*)d_in[0];
  const float* Wih = (const float*)d_in[1];
  const float* Whh = (const float*)d_in[2];
  const float* bih = (const float*)d_in[3];
  const float* bhh = (const float*)d_in[4];
  const float* Adw = (const float*)d_in[5];
  const float* Adb = (const float*)d_in[6];
  const float* Auw = (const float*)d_in[7];
  const float* Aub = (const float*)d_in[8];
  const float* lng = (const float*)d_in[9];
  const float* lnb = (const float*)d_in[10];
  float* out   = (float*)d_out;
  float* out_x = out;
  float* out_h = out + (size_t)T_ * B_ * H_;
  float* out_c = out_h + (size_t)B_ * H_;

  // workspace carve-up (~386 MB)
  char* ws = (char*)d_ws;
  size_t off = 0;
  auto alloc = [&](size_t bytes) { char* p = ws + off; off += (bytes + 255) & ~(size_t)255; return p; };
  float*          gxb   = (float*)alloc((size_t)T_ * B_ * G4H * 4);     // 256 MB
  float*          act   = (float*)alloc((size_t)T_ * B_ * H_ * 4);      //  64 MB
  unsigned short* xa    = (unsigned short*)alloc((size_t)T_ * B_ * H_ * 2); // 32 MB
  unsigned short* wih_s = (unsigned short*)alloc((size_t)G4H * H_ * 2); //   8 MB
  unsigned short* whh_s = (unsigned short*)alloc((size_t)G4H * H_ * 2); //   8 MB
  float*          gbias = (float*)alloc((size_t)G4H * 4);
  unsigned short* hbf   = (unsigned short*)alloc((size_t)2 * B_ * H_ * 2); // ping-pong h
  float*          h_st  = (float*)alloc((size_t)B_ * H_ * 4);
  float*          c_st  = (float*)alloc((size_t)B_ * H_ * 4);
  unsigned*       ctr   = (unsigned*)alloc(256);

  (void)hipMemsetAsync(h_st, 0, (size_t)B_ * H_ * 4, stream);   // h0 = 0
  (void)hipMemsetAsync(c_st, 0, (size_t)B_ * H_ * 4, stream);   // c0 = 0

  const int TBH = T_ * B_ * H_;
  k_f32_to_bf16<<<(TBH + 255) / 256, 256, 0, stream>>>(x, xa, TBH);

  const size_t smem = (size_t)B_ * H_ * 2 + (size_t)B_ * 128 * 4 + (size_t)B_ * 32 * 4; // 84 KB

  for (int l = 0; l < 2; ++l) {
    k_swizzle_w<<<(G4H * H_ / 2) / 256, 256, 0, stream>>>(Wih + (size_t)l * G4H * H_, wih_s);
    k_swizzle_w<<<(G4H * H_ / 2) / 256, 256, 0, stream>>>(Whh + (size_t)l * G4H * H_, whh_s);
    k_bias<<<G4H / 256, 256, 0, stream>>>(bih + l * G4H, bhh + l * G4H, gbias);

    // gx = x @ Wih^T + bias   (M=16384, N=4096, K=1024)
    k_gemm_gx<<<dim3(G4H / 256, (T_ * B_) / 64), 256, 0, stream>>>(xa, wih_s, gbias, gxb);

    // h0 -> bf16 slab 0; reset barrier counter
    k_f32_to_bf16<<<(B_ * H_ + 255) / 256, 256, 0, stream>>>(h_st, hbf, B_ * H_);
    (void)hipMemsetAsync(ctr, 0, 4, stream);

    float* h_o = (l == 1) ? out_h : h_st;
    float* c_o = (l == 1) ? out_c : c_st;
    k_scan<<<NWG, 256, smem, stream>>>(gxb, whh_s, hbf, c_st, act, h_o, c_o, ctr);

    float* xo = (l == 1) ? out_x : act;   // layer0: in-place; layer1: final output
    k_adapter_ln<<<T_ * B_, 256, 0, stream>>>(act,
        Adw + (size_t)l * DA_ * H_, Adb + (size_t)l * DA_,
        Auw + (size_t)l * H_ * DA_, Aub + (size_t)l * H_,
        lng + (size_t)l * H_, lnb + (size_t)l * H_, xo, xa);
  }
}
